// GridAttention_11819749998707
// MI455X (gfx1250) — matrix operand order repaired and checked by reading, not compiled, not run on hardware
//
#include <hip/hip_runtime.h>
#include <hip/hip_bf16.h>

// ---------------------------------------------------------------------------
// GridAttention (MaxViT grid attention w/ Swin-V2 cosine attn + CPB) for
// gfx1250 (MI455X).  All heavy GEMMs + both attention GEMMs use
// v_wmma_f32_16x16x32_bf16 (fp32 accumulate).  Intermediates held in bf16 in
// workspace to halve HBM traffic (op is bandwidth bound: ~0.8GB @ 23.3TB/s).
// GEMM uses double-buffered LDS staging and coalesced b128 epilogue stores.
// ---------------------------------------------------------------------------

#define HEADS   16
#define C_      512
#define L_      64
#define G_      1024
#define M_TOK   65536
#define NQKV    1536
#define LOG100  4.6051701859880914f

typedef __bf16 bf16_t;
typedef __attribute__((ext_vector_type(16))) bf16_t v16bf;
typedef __attribute__((ext_vector_type(8)))  bf16_t v8bf;
typedef __attribute__((ext_vector_type(8)))  float  v8f;

// workspace layout (bytes)
#define OFF_RBIAS 0u           // 16*64*64 f32  = 262144
#define OFF_B225  262144u      // 225*16 f32
#define OFF_FBIAS 278528u      // 1536 f32
#define OFF_QKV   286720u      // 65536*1536 bf16 = 201326592
#define OFF_ATTN  201613312ull // 65536*512  bf16 = 67108864

// ---------------------------------------------------------------------------
// fused qkv bias: [q_bias, zeros, v_bias]
__global__ void prep_fbias_kernel(const float* __restrict__ qb,
                                  const float* __restrict__ vb,
                                  float* __restrict__ fb) {
  int i = blockIdx.x * blockDim.x + threadIdx.x;
  if (i < NQKV) fb[i] = (i < 512) ? qb[i] : (i < 1024 ? 0.f : vb[i - 1024]);
}

// ---------------------------------------------------------------------------
// CPB MLP: table[225,2] -> relu(512) -> heads(16).  One thread per table row.
__global__ void cpb_mlp_kernel(const float* __restrict__ w1,
                               const float* __restrict__ b1,
                               const float* __restrict__ w2,
                               float* __restrict__ b225) {
  int r = threadIdx.x;
  if (r >= 225) return;
  int i = r / 15, j = r % 15;
  float ti = (float)(i - 7) * (8.0f / 7.0f);
  float tj = (float)(j - 7) * (8.0f / 7.0f);
  const float inv_l2_8 = 1.0f / 3.0f; // 1/log2(8)
  float sgi = (ti > 0.f) ? 1.f : ((ti < 0.f) ? -1.f : 0.f);
  float sgj = (tj > 0.f) ? 1.f : ((tj < 0.f) ? -1.f : 0.f);
  float t0 = sgi * log2f(fabsf(ti) + 1.f) * inv_l2_8;
  float t1 = sgj * log2f(fabsf(tj) + 1.f) * inv_l2_8;
  float acc[HEADS];
#pragma unroll
  for (int h = 0; h < HEADS; ++h) acc[h] = 0.f;
  for (int jj = 0; jj < 512; ++jj) {
    float hm = fmaxf(t0 * w1[jj] + t1 * w1[512 + jj] + b1[jj], 0.f);
#pragma unroll
    for (int h = 0; h < HEADS; ++h) acc[h] += hm * w2[jj * HEADS + h];
  }
#pragma unroll
  for (int h = 0; h < HEADS; ++h) b225[r * HEADS + h] = acc[h];
}

// scatter through relative index, 16*sigmoid -> rbias[head][l][m]
__global__ void cpb_scatter_kernel(const float* __restrict__ b225,
                                   float* __restrict__ rbias) {
  int idx = blockIdx.x * blockDim.x + threadIdx.x;
  if (idx >= HEADS * L_ * L_) return;
  int h = idx >> 12;
  int l = (idx >> 6) & 63;
  int m = idx & 63;
  int r = ((l >> 3) - (m >> 3) + 7) * 15 + ((l & 7) - (m & 7) + 7);
  float x = b225[r * HEADS + h];
  rbias[idx] = 16.f / (1.f + __expf(-x));
}

// ---------------------------------------------------------------------------
// Generic bf16 WMMA GEMM: C[M,N] = A[M,K] * B[K,N] + bias[N]
// block tile 128x128, BK=32, 256 threads (8 waves -> 2x4 wave grid, 64x32
// each).  Double-buffered LDS; bf16 epilogue re-staged through LDS (aliasing
// the dead A/B buffers) for coalesced b128 global stores.
template <bool A_BF16, bool OUT_F32>
__global__ __launch_bounds__(256) void gemm_bf16_kernel(
    const void* __restrict__ Aptr, const float* __restrict__ Bw,
    const float* __restrict__ bias, void* __restrict__ Outp,
    int M, int N, int K) {
  (void)M;
  // 2 x (As[128][40] + Bs[128][40]) bf16 = 40960 B; Ct[128][136] aliases it.
  __shared__ __align__(32) bf16_t smem[2 * 128 * 40 + 2 * 128 * 40];
  typedef bf16_t(*TileT)[128][40];
  TileT As = (TileT)smem;
  TileT Bs = (TileT)(smem + 2 * 128 * 40);
  typedef bf16_t(*CtT)[136];
  CtT Ct = (CtT)smem; // valid only after the K loop's final barrier

  const int tid  = threadIdx.x;
  const int lane = tid & 31;
  const int wave = tid >> 5;
  const int wm   = wave >> 2; // 0..1
  const int wn   = wave & 3;  // 0..3
  const int ln   = lane & 15;
  const int hf   = lane >> 4;
  const int m0   = blockIdx.x * 128;
  const int n0   = blockIdx.y * 128;

  v8f acc[4][2] = {};

  const int arow = tid >> 1, aseg = (tid & 1) * 16;  // A: 2 thr/row, 16 elem
  const int bk = tid >> 3, bn = (tid & 7) * 16;      // B: 8 thr/row, 16 elem

  // prefetch registers
  float4 pf0, pf1, pf2, pf3; // A fp32
  v8bf pa0, pa1;             // A bf16
  float4 pg0, pg1, pg2, pg3; // B fp32

  auto fetchA = [&](int kb) {
    if (A_BF16) {
      const bf16_t* A = (const bf16_t*)Aptr;
      const v8bf* src = (const v8bf*)(A + (size_t)(m0 + arow) * K + kb + aseg);
      pa0 = src[0];
      pa1 = src[1];
    } else {
      const float* A = (const float*)Aptr;
      const float4* src =
          (const float4*)(A + (size_t)(m0 + arow) * K + kb + aseg);
      pf0 = src[0]; pf1 = src[1]; pf2 = src[2]; pf3 = src[3];
    }
  };
  auto storeA = [&](int buf) {
    if (A_BF16) {
      *(v8bf*)&As[buf][arow][aseg]     = pa0;
      *(v8bf*)&As[buf][arow][aseg + 8] = pa1;
    } else {
      v8bf o0, o1;
      o0[0] = (bf16_t)pf0.x; o0[1] = (bf16_t)pf0.y; o0[2] = (bf16_t)pf0.z; o0[3] = (bf16_t)pf0.w;
      o0[4] = (bf16_t)pf1.x; o0[5] = (bf16_t)pf1.y; o0[6] = (bf16_t)pf1.z; o0[7] = (bf16_t)pf1.w;
      o1[0] = (bf16_t)pf2.x; o1[1] = (bf16_t)pf2.y; o1[2] = (bf16_t)pf2.z; o1[3] = (bf16_t)pf2.w;
      o1[4] = (bf16_t)pf3.x; o1[5] = (bf16_t)pf3.y; o1[6] = (bf16_t)pf3.z; o1[7] = (bf16_t)pf3.w;
      *(v8bf*)&As[buf][arow][aseg]     = o0;
      *(v8bf*)&As[buf][arow][aseg + 8] = o1;
    }
  };
  auto fetchB = [&](int kb) {
    const float4* src = (const float4*)(Bw + (size_t)(kb + bk) * N + n0 + bn);
    pg0 = src[0]; pg1 = src[1]; pg2 = src[2]; pg3 = src[3];
  };
  auto storeB = [&](int buf) {
    float tmp[16] = {pg0.x, pg0.y, pg0.z, pg0.w, pg1.x, pg1.y, pg1.z, pg1.w,
                     pg2.x, pg2.y, pg2.z, pg2.w, pg3.x, pg3.y, pg3.z, pg3.w};
#pragma unroll
    for (int e = 0; e < 16; ++e) Bs[buf][bn + e][bk] = (bf16_t)tmp[e];
  };

  // prologue: stage tile 0
  fetchA(0);
  fetchB(0);
  storeA(0);
  storeB(0);
  __syncthreads();

  const int nK = K >> 5;
  for (int ki = 0; ki < nK; ++ki) {
    const int cur = ki & 1;
    const bool has_next = (ki + 1) < nK;
    if (has_next) { // issue next-tile global loads before compute
      fetchA((ki + 1) << 5);
      fetchB((ki + 1) << 5);
    }
    // ---- WMMA on current buffer: 4 m-tiles x 2 n-tiles, K=32 each ----
    v16bf afr[4];
#pragma unroll
    for (int mt = 0; mt < 4; ++mt) {
      int m = wm * 64 + mt * 16 + ln;
      v8bf* ah = (v8bf*)&afr[mt];
      ah[0] = *(const v8bf*)&As[cur][m][hf * 8];
      ah[1] = *(const v8bf*)&As[cur][m][16 + hf * 8];
    }
    v16bf bfr[2];
#pragma unroll
    for (int nt = 0; nt < 2; ++nt) {
      int n = wn * 32 + nt * 16 + ln;
      v8bf* bh = (v8bf*)&bfr[nt];
      bh[0] = *(const v8bf*)&Bs[cur][n][hf * 8];
      bh[1] = *(const v8bf*)&Bs[cur][n][16 + hf * 8];
    }
#pragma unroll
    for (int mt = 0; mt < 4; ++mt)
#pragma unroll
      for (int nt = 0; nt < 2; ++nt)
        acc[mt][nt] = __builtin_amdgcn_wmma_f32_16x16x32_bf16(
            false, afr[mt], false, bfr[nt], (short)0, acc[mt][nt], false,
            false);
    if (has_next) { // store prefetched tile to alternate buffer
      storeA(cur ^ 1);
      storeB(cur ^ 1);
    }
    __syncthreads();
  }

  // ---- epilogue ----
  if (OUT_F32) {
    // fp32 out: direct coalesced b32 stores w/ non-temporal hint (134MB,
    // never re-read -> keep L2 for the bf16 working set).
#pragma unroll
    for (int mt = 0; mt < 4; ++mt) {
#pragma unroll
      for (int nt = 0; nt < 2; ++nt) {
        int col  = n0 + wn * 32 + nt * 16 + ln;
        float bv = bias[col];
#pragma unroll
        for (int e = 0; e < 8; ++e) {
          int row = m0 + wm * 64 + mt * 16 + hf * 8 + e;
          float v = acc[mt][nt][e] + bv;
          __builtin_nontemporal_store(v, &((float*)Outp)[(size_t)row * N + col]);
        }
      }
    }
  } else {
    // bf16 out: transpose-stage through LDS (aliases dead A/B buffers),
    // then fully coalesced 16B (v8bf) global stores.
#pragma unroll
    for (int mt = 0; mt < 4; ++mt) {
#pragma unroll
      for (int nt = 0; nt < 2; ++nt) {
        int cc  = wn * 32 + nt * 16 + ln;
        float bv = bias[n0 + cc];
#pragma unroll
        for (int e = 0; e < 8; ++e) {
          int rr = wm * 64 + mt * 16 + hf * 8 + e;
          Ct[rr][cc] = (bf16_t)(acc[mt][nt][e] + bv);
        }
      }
    }
    __syncthreads();
    const int row = tid >> 1, seg = (tid & 1) * 64;
    const bf16_t* srcr = &Ct[row][seg];
    bf16_t* dstr = (bf16_t*)Outp + (size_t)(m0 + row) * N + n0 + seg;
#pragma unroll
    for (int c = 0; c < 8; ++c)
      *(v8bf*)(dstr + c * 8) = *(const v8bf*)(srcr + c * 8);
  }
}

// ---------------------------------------------------------------------------
// Fused grid attention: one block per (group, head). 4 waves; wave w owns
// query rows [16w,16w+16).  S = (l2n(q)*temp)·l2n(k)^T + rbias ; softmax ;
// O = P·V.  All matmuls are v_wmma_f32_16x16x32_bf16.
__global__ __launch_bounds__(128) void grid_attn_kernel(
    const bf16_t* __restrict__ qkv, const float* __restrict__ rbias,
    const float* __restrict__ scale, bf16_t* __restrict__ aout) {
  __shared__ bf16_t qh[64][32];
  __shared__ bf16_t kh[64][32];
  __shared__ bf16_t vv[64][32];
  __shared__ bf16_t pp[4][16][64];

  const int g = blockIdx.x, head = blockIdx.y;
  const int tid = threadIdx.x;
  const int b = g >> 8, hb = (g >> 4) & 15, wb = g & 15;

  auto tok = [&](int l) {
    int wi = l >> 3, wj = l & 7;
    return (b * 128 + wi * 16 + hb) * 128 + wj * 16 + wb;
  };

  // ---- load + l2-normalize q (threads 0-63) / k (threads 64-127) ----
  {
    int l = tid & 63;
    int sel = tid >> 6; // 0:q  1:k
    int t = tok(l);
    const v8bf* sp =
        (const v8bf*)(qkv + (size_t)t * NQKV + sel * 512 + head * 32);
    float f[32];
#pragma unroll
    for (int c = 0; c < 4; ++c) {
      v8bf x = sp[c];
#pragma unroll
      for (int e = 0; e < 8; ++e) f[c * 8 + e] = (float)x[e];
    }
    float ss = 0.f;
#pragma unroll
    for (int u = 0; u < 32; ++u) ss += f[u] * f[u];
    float r = rsqrtf(fmaxf(ss, 1.55e-5f));
    bf16_t* dst = sel ? &kh[l][0] : &qh[l][0];
#pragma unroll
    for (int u = 0; u < 32; ++u) dst[u] = (bf16_t)(f[u] * r);
  }
  // ---- load v (all threads, half row each) ----
  {
    int l = tid & 63, half = tid >> 6;
    int t = tok(l);
    const v8bf* sp =
        (const v8bf*)(qkv + (size_t)t * NQKV + 1024 + head * 32 + half * 16);
    *(v8bf*)&vv[l][half * 16]     = sp[0];
    *(v8bf*)&vv[l][half * 16 + 8] = sp[1];
  }
  __syncthreads();

  const int lane = tid & 31, wave = tid >> 5;
  const int ln = lane & 15, hf = lane >> 4;
  const float temp = __expf(fminf(scale[head], LOG100));

  // A-operand: this wave's 16 q rows
  v16bf aq;
  {
    int m = wave * 16 + ln;
    v8bf* ah = (v8bf*)&aq;
    ah[0] = *(const v8bf*)&qh[m][hf * 8];
    ah[1] = *(const v8bf*)&qh[m][16 + hf * 8];
  }

  // ---- S = q·k^T  (4 WMMAs, K=32 each), scale + rbias ----
  float sv[4][8];
  const float* rb = rbias + head * 4096;
#pragma unroll
  for (int nt = 0; nt < 4; ++nt) {
    v16bf bk;
    {
      int n = nt * 16 + ln;
      v8bf* bh = (v8bf*)&bk;
      bh[0] = *(const v8bf*)&kh[n][hf * 8];
      bh[1] = *(const v8bf*)&kh[n][16 + hf * 8];
    }
    v8f z = {};
    v8f s = __builtin_amdgcn_wmma_f32_16x16x32_bf16(false, aq, false, bk,
                                                    (short)0, z, false, false);
    int mrow = wave * 16 + hf * 8;
#pragma unroll
    for (int e = 0; e < 8; ++e)
      sv[nt][e] = s[e] * temp + rb[(mrow + e) * 64 + nt * 16 + ln];
  }

  // ---- row softmax (reduce across 16-lane half + 4 n-tiles) ----
  float inv[8];
#pragma unroll
  for (int e = 0; e < 8; ++e) {
    float mx = fmaxf(fmaxf(sv[0][e], sv[1][e]), fmaxf(sv[2][e], sv[3][e]));
#pragma unroll
    for (int msk = 1; msk < 16; msk <<= 1)
      mx = fmaxf(mx, __shfl_xor(mx, msk, 32));
    float sm = 0.f;
#pragma unroll
    for (int nt = 0; nt < 4; ++nt) {
      float p = __expf(sv[nt][e] - mx);
      sv[nt][e] = p;
      sm += p;
    }
#pragma unroll
    for (int msk = 1; msk < 16; msk <<= 1) sm += __shfl_xor(sm, msk, 32);
    inv[e] = 1.f / sm;
  }

  // ---- stage P (bf16) through LDS to re-layout as A operand ----
#pragma unroll
  for (int nt = 0; nt < 4; ++nt)
#pragma unroll
    for (int e = 0; e < 8; ++e)
      pp[wave][hf * 8 + e][nt * 16 + ln] = (bf16_t)(sv[nt][e] * inv[e]);
  __syncthreads();

  // ---- O = P·V  (2 K-steps x 2 n-tiles = 4 WMMAs) ----
  v8f oacc[2] = {};
#pragma unroll
  for (int ks = 0; ks < 2; ++ks) {
    v16bf ap;
    {
      v8bf* ah = (v8bf*)&ap;
      ah[0] = *(const v8bf*)&pp[wave][ln][ks * 32 + hf * 8];
      ah[1] = *(const v8bf*)&pp[wave][ln][ks * 32 + 16 + hf * 8];
    }
#pragma unroll
    for (int nv = 0; nv < 2; ++nv) {
      v16bf bv;
      int u = nv * 16 + ln;
      int k0 = ks * 32 + hf * 8;
#pragma unroll
      for (int e = 0; e < 16; ++e) bv[e] = vv[k0 + e + (e & 8)][u];
      oacc[nv] = __builtin_amdgcn_wmma_f32_16x16x32_bf16(
          false, ap, false, bv, (short)0, oacc[nv], false, false);
    }
  }

  // ---- store token-major bf16 [65536][512] ----
#pragma unroll
  for (int nv = 0; nv < 2; ++nv) {
    int col = head * 32 + nv * 16 + ln;
#pragma unroll
    for (int e = 0; e < 8; ++e) {
      int l = wave * 16 + hf * 8 + e;
      int t = tok(l);
      aout[(size_t)t * 512 + col] = (bf16_t)oacc[nv][e];
    }
  }
}

// ---------------------------------------------------------------------------
extern "C" void kernel_launch(void* const* d_in, const int* in_sizes, int n_in,
                              void* d_out, int out_size, void* d_ws,
                              size_t ws_size, hipStream_t stream) {
  (void)in_sizes; (void)n_in; (void)out_size; (void)ws_size;
  const float* x      = (const float*)d_in[0];
  const float* qkv_w  = (const float*)d_in[1];
  const float* q_bias = (const float*)d_in[2];
  const float* v_bias = (const float*)d_in[3];
  const float* scale  = (const float*)d_in[4];
  const float* cpb_w1 = (const float*)d_in[5];
  const float* cpb_b1 = (const float*)d_in[6];
  const float* cpb_w2 = (const float*)d_in[7];
  const float* proj_w = (const float*)d_in[8];
  const float* proj_b = (const float*)d_in[9];
  float* out = (float*)d_out;

  char* ws = (char*)d_ws;
  float*  rbias = (float*)(ws + OFF_RBIAS);
  float*  b225  = (float*)(ws + OFF_B225);
  float*  fbias = (float*)(ws + OFF_FBIAS);
  bf16_t* qkv   = (bf16_t*)(ws + OFF_QKV);
  bf16_t* attn  = (bf16_t*)(ws + OFF_ATTN);

  prep_fbias_kernel<<<dim3(6), dim3(256), 0, stream>>>(q_bias, v_bias, fbias);
  cpb_mlp_kernel<<<dim3(1), dim3(256), 0, stream>>>(cpb_w1, cpb_b1, cpb_w2,
                                                    b225);
  cpb_scatter_kernel<<<dim3(256), dim3(256), 0, stream>>>(b225, rbias);

  // qkv = x @ qkv_w + [q_bias,0,v_bias]   (bf16 out)
  gemm_bf16_kernel<false, false><<<dim3(512, 12), dim3(256), 0, stream>>>(
      (const void*)x, qkv_w, fbias, (void*)qkv, M_TOK, NQKV, 512);

  // fused grid attention per (group, head)
  grid_attn_kernel<<<dim3(G_, HEADS), dim3(128), 0, stream>>>(qkv, rbias,
                                                              scale, attn);

  // out = attn @ proj_w + proj_b   (fp32 out)
  gemm_bf16_kernel<true, true><<<dim3(512, 4), dim3(256), 0, stream>>>(
      (const void*)attn, proj_w, proj_b, (void*)out, M_TOK, 512, 512);
}
